// SFF_61100204753550
// MI455X (gfx1250) — matrix operand order repaired and checked by reading, not compiled, not run on hardware
//
#include <hip/hip_runtime.h>

// ---------------------------------------------------------------------------
// Problem constants (B=8, C=128, H=W=128)
#define Bn    8
#define Cc    128
#define C2    256
#define HH    128
#define WW    128
#define HWSZ  16384          // H*W
#define NSP   131072         // B*H*W
#define KCONV 2304           // 2C * 9

typedef __attribute__((ext_vector_type(16))) __bf16 v16bf;
typedef __attribute__((ext_vector_type(8)))  float  v8f;
typedef __attribute__((ext_vector_type(4)))  unsigned int v4u;
typedef __attribute__((ext_vector_type(8)))  int  v8i;
typedef __attribute__((ext_vector_type(4)))  int  v4i;

union ABfrag { v16bf v; unsigned int u[8]; };

__device__ __forceinline__ unsigned short f2bf(float f) {
    unsigned int u = __float_as_uint(f);
    unsigned int r = (u + 0x7FFFu + ((u >> 16) & 1u)) >> 16;   // RNE
    return (unsigned short)r;
}

// ---------------------------------------------------------------------------
// K1: per-(b,channel) global mean over H*W of concatenated [x1;x2]
__global__ void k_mean(const float* __restrict__ x1, const float* __restrict__ x2,
                       float* __restrict__ avg) {
    int bc = blockIdx.x;                 // 0..2047 = b*256 + c
    int b = bc >> 8, c = bc & 255;
    const float* src = (c < Cc) ? x1 + (((size_t)b * Cc + c) << 14)
                                : x2 + (((size_t)b * Cc + (c - Cc)) << 14);
    float s = 0.f;
    for (int i = threadIdx.x; i < HWSZ; i += 256) s += src[i];
    __shared__ float red[256];
    red[threadIdx.x] = s; __syncthreads();
    for (int off = 128; off > 0; off >>= 1) {
        if (threadIdx.x < off) red[threadIdx.x] += red[threadIdx.x + off];
        __syncthreads();
    }
    if (threadIdx.x == 0) avg[bc] = red[0] * (1.f / (float)HWSZ);
}

// ---------------------------------------------------------------------------
// K2: SE MLP (relu(avg@W1^T) -> sigmoid(@W2^T)) + BN inv/shift precompute
__global__ void k_se(const float* __restrict__ avg,
                     const float* __restrict__ w1, const float* __restrict__ w2,
                     const float* __restrict__ g,  const float* __restrict__ bta,
                     const float* __restrict__ mean, const float* __restrict__ var,
                     float* __restrict__ se, float* __restrict__ bnp) {
    __shared__ float abuf[Bn * C2];
    __shared__ float hbuf[Bn * 32];
    int t = threadIdx.x;
    for (int i = t; i < Bn * C2; i += 256) abuf[i] = avg[i];
    __syncthreads();
    {   // h[8][32]
        int i = t >> 5, j = t & 31;
        const float* wr = w1 + j * C2;
        float s = 0.f;
        for (int k = 0; k < C2; ++k) s += abuf[i * C2 + k] * wr[k];
        hbuf[t] = fmaxf(s, 0.f);
    }
    __syncthreads();
    for (int idx = t; idx < Bn * C2; idx += 256) {   // se[8][256]
        int i = idx >> 8, c = idx & 255;
        const float* wr = w2 + c * 32;
        float s = 0.f;
        for (int k = 0; k < 32; ++k) s += hbuf[i * 32 + k] * wr[k];
        se[idx] = 1.f / (1.f + __expf(-s));
    }
    if (t < Cc) {
        float inv = g[t] * rsqrtf(var[t] + 1e-5f);
        bnp[t] = inv;
        bnp[Cc + t] = bta[t] - mean[t] * inv;
    }
}

// ---------------------------------------------------------------------------
// K3: xh[b][ci][h][w] = bf16( x * se[b][ci] ), ci in [0,256)
__global__ void k_scale(const float* __restrict__ x1, const float* __restrict__ x2,
                        const float* __restrict__ se, unsigned short* __restrict__ xh) {
    size_t i4 = ((((size_t)blockIdx.x << 8) + threadIdx.x) << 2);
    size_t bc = i4 >> 14;
    int b = (int)(bc >> 8), c = (int)(bc & 255);
    const float* src = (c < Cc) ? x1 + (((size_t)b * Cc + c) << 14)
                                : x2 + (((size_t)b * Cc + (c - Cc)) << 14);
    int off = (int)(i4 & 16383);
    float4 v = *(const float4*)(src + off);
    float s = se[bc];
    unsigned int lo = (unsigned int)f2bf(v.x * s) | ((unsigned int)f2bf(v.y * s) << 16);
    unsigned int hi = (unsigned int)f2bf(v.z * s) | ((unsigned int)f2bf(v.w * s) << 16);
    *(uint2*)(xh + i4) = make_uint2(lo, hi);
}

// K3b: conv weights f32 -> bf16 repacked as A[co][k2], k2 = (ky*3+kx)*256 + ci.
// This K-ordering makes the conv fill's filter tap uniform per K-slab.
__global__ void k_wconv(const float* __restrict__ w, unsigned short* __restrict__ wbf) {
    int i  = (blockIdx.x << 8) + threadIdx.x;     // 73728 threads, 4 outputs each
    int o4 = i << 2;
    int co  = o4 / KCONV;
    int rem = o4 - co * KCONV;                    // r*256 + ci  (ci fastest, mult of 4)
    int r   = rem >> 8;
    int ci  = rem & 255;
    const float* src = w + (size_t)co * KCONV + r;   // + ci*9
    unsigned int lo = (unsigned int)f2bf(src[(size_t)(ci + 0) * 9])
                    | ((unsigned int)f2bf(src[(size_t)(ci + 1) * 9]) << 16);
    unsigned int hi = (unsigned int)f2bf(src[(size_t)(ci + 2) * 9])
                    | ((unsigned int)f2bf(src[(size_t)(ci + 3) * 9]) << 16);
    *(uint2*)(wbf + o4) = make_uint2(lo, hi);
}

// ---------------------------------------------------------------------------
// K4: 3x3 conv as implicit GEMM with v_wmma_f32_16x16x32_bf16.
// K axis ordered tap-major (k2 = r*256 + ci): dy/dx and the pad predicate are
// hoisted; the fill is branchless (clamped address + keep-mask), packed into
// two ds_store_b32 per thread per K-slab.
__global__ void k_conv(const unsigned short* __restrict__ xh,
                       const unsigned short* __restrict__ wbf,
                       const float* __restrict__ bnp,
                       float* __restrict__ y, unsigned short* __restrict__ qb) {
    __shared__ unsigned short Bl[32 * 34];   // [n][k], row pad 2 -> 4B aligned K-pairs
    __shared__ float bnL[2 * Cc];
    int t = threadIdx.x;
    bnL[t] = bnp[t];
    int lane = t & 31, wv = t >> 5;
    int coB = wv << 4;
    int nBase = blockIdx.x << 5;             // 32 spatial cols (same batch & image row)
    int b = nBase >> 14;
    const unsigned short* xb = xh + ((size_t)b * C2 << 14);

    // fill roles: each warp handles one k-quarter (4 ci's) for all 32 n's
    int fn = t & 31, fkq = t >> 5;
    int hw = (nBase + fn) & 16383;
    int hh = hw >> 7, w0 = hw & 127;
    unsigned short* Bf = Bl + fn * 34 + (fkq << 2);   // 4B aligned

    v8f zero = {};
    v8f acc0 = zero, acc1 = zero;

    const unsigned short* arow0 =
        wbf + (size_t)(coB + (lane & 15)) * KCONV + ((lane >> 4) << 4);
    const unsigned short* p0 = Bl + (lane & 15) * 34 + ((lane >> 4) << 4);
    const unsigned short* p1 = p0 + 16 * 34;

    for (int r9 = 0; r9 < 9; ++r9) {
        int dy = r9 / 3 - 1;
        int dx = r9 - (r9 / 3) * 3 - 1;
        int yy = hh + dy, xx = w0 + dx;
        bool ok = ((unsigned)yy < 128u) & ((unsigned)xx < 128u);
        unsigned int keepMask = ok ? 0xFFFFFFFFu : 0u;           // hoisted select
        int yc = min(max(yy, 0), 127), xc = min(max(xx, 0), 127); // always-valid addr
        long spOff = ((long)(fkq << 2) << 14) + ((long)yc << 7) + xc;  // + ci<<14 later

        for (int cb = 0; cb < 256; cb += 32) {
            int kk = (r9 << 8) + cb;
            __syncthreads();
            {   // branchless im2col fill: 4 loads -> 2 packed dword LDS stores
                unsigned int a0 = xb[spOff + ((long)(cb + 0) << 14)];
                unsigned int a1 = xb[spOff + ((long)(cb + 1) << 14)];
                unsigned int a2 = xb[spOff + ((long)(cb + 2) << 14)];
                unsigned int a3 = xb[spOff + ((long)(cb + 3) << 14)];
                *(unsigned int*)(Bf)     = (a0 | (a1 << 16)) & keepMask;
                *(unsigned int*)(Bf + 2) = (a2 | (a3 << 16)) & keepMask;
            }
            __syncthreads();

            ABfrag a;
            {   // A fragment, contiguous k = 16*(lane/16) + s, same as B (2x b128 from global)
                const unsigned short* arow = arow0 + kk;
                __builtin_prefetch(arow + 64, 0, 0);   // global_prefetch_b8
                #pragma unroll
                for (int v = 0; v < 8; ++v)
                    a.u[v] = *(const unsigned int*)(arow + (v << 1));
            }
            ABfrag b0, b1;
            #pragma unroll
            for (int v = 0; v < 8; ++v) {
                b0.u[v] = *(const unsigned int*)(p0 + (v << 1));
                b1.u[v] = *(const unsigned int*)(p1 + (v << 1));
            }
            acc0 = __builtin_amdgcn_wmma_f32_16x16x32_bf16(false, a.v, false, b0.v,
                                                           (short)0, acc0, false, false);
            acc1 = __builtin_amdgcn_wmma_f32_16x16x32_bf16(false, a.v, false, b1.v,
                                                           (short)0, acc1, false, false);
        }
    }

    // BN + ReLU epilogue; C/D layout: VGPR r -> M = r + 8*(lane>=16), N = lane%16
    int hw0 = (nBase & 16383) + (lane & 15);
    #pragma unroll
    for (int r = 0; r < 8; ++r) {
        int co = coB + r + ((lane >> 4) << 3);
        float inv = bnL[co], sh = bnL[Cc + co];
        float v0 = fmaxf(acc0[r] * inv + sh, 0.f);
        float v1 = fmaxf(acc1[r] * inv + sh, 0.f);
        size_t i0 = (((size_t)(b * Cc + co)) << 14) + hw0;
        y[i0]       = v0;  y[i0 + 16]  = v1;
        qb[i0]      = f2bf(v0);
        qb[i0 + 16] = f2bf(v1);
    }
}

// ---------------------------------------------------------------------------
// K5: per-batch Gram G = q q^T (128x128, K=16384) + softmax(rowmin - G) -> sim bf16.
// K-slab of q staged in LDS via the Tensor Data Mover (2D tile 32x128, padded
// rows via D# pad fields), reused as both the A and B fragment source.
__global__ void k_gram(const unsigned short* __restrict__ qb,
                       unsigned short* __restrict__ simb) {
    __shared__ unsigned short Qs[128 * 34];
    int t = threadIdx.x, lane = t & 31, wv = t >> 5;
    int b = blockIdx.x;
    const unsigned short* q = qb + ((size_t)b << 21);   // 128*16384

    v8f zero = {};
    v8f acc[8];
    #pragma unroll
    for (int j = 0; j < 8; ++j) acc[j] = zero;

#if __has_builtin(__builtin_amdgcn_tensor_load_to_lds)
    // D# group1: data_size=2B, pad_enable, pad_interval=16 DW, pad_amount=1 DW
    // tensor_dim0=16384, tensor_dim1=128, tile_dim0=32, tile_dim1=128,
    // tensor_dim0_stride=16384  -> LDS row stride 68 B (matches Qs[128][34])
    unsigned int ldsOff = (unsigned int)(unsigned long long)(const void*)Qs;
    v8i g1 = { (int)0x00D10000, (int)0x40000000, (int)0x00800000, (int)0x00200000,
               128, 16384, 0, 0 };
    v4i gz4 = { 0, 0, 0, 0 };
    v8i gz8 = { 0, 0, 0, 0, 0, 0, 0, 0 };
#else
    int frow = t >> 1, fhalf = t & 1;
#endif

    for (int kk = 0; kk < HWSZ; kk += 32) {
        __syncthreads();
#if __has_builtin(__builtin_amdgcn_tensor_load_to_lds)
        if (wv == 0) {   // one wave issues the TDM DMA (EXEC ignored by TDM)
            unsigned long long ga = (unsigned long long)(const void*)(q + kk);
            v4u g0;
            g0.x = 1u;                                   // count=1, user mode
            g0.y = ldsOff;                               // lds_addr
            g0.z = (unsigned int)(ga & 0xffffffffu);     // global_addr[31:0]
            g0.w = (unsigned int)((ga >> 32) & 0x01FFFFFFu) | 0x80000000u; // +type=2
            __builtin_amdgcn_tensor_load_to_lds(g0, g1, gz4, gz4, gz8, 0);
            __builtin_amdgcn_s_wait_tensorcnt(0);
        }
#else
        {   // fallback: manual staged copy, padded rows
            uint4 gv = *(const uint4*)(q + (size_t)frow * HWSZ + kk + fhalf * 16);
            unsigned int* dst = (unsigned int*)(Qs + frow * 34 + fhalf * 16);
            dst[0] = gv.x; dst[1] = gv.y; dst[2] = gv.z; dst[3] = gv.w;
        }
#endif
        __syncthreads();

        ABfrag a;
        {
            const unsigned short* ar = Qs + (wv * 16 + (lane & 15)) * 34 + ((lane >> 4) << 4);
            #pragma unroll
            for (int v = 0; v < 8; ++v)
                a.u[v] = *(const unsigned int*)(ar + (v << 1));
        }
        #pragma unroll
        for (int j = 0; j < 8; ++j) {
            ABfrag bb;
            const unsigned short* br = Qs + (j * 16 + (lane & 15)) * 34 + ((lane >> 4) << 4);
            #pragma unroll
            for (int v = 0; v < 8; ++v)
                bb.u[v] = *(const unsigned int*)(br + (v << 1));
            acc[j] = __builtin_amdgcn_wmma_f32_16x16x32_bf16(false, a.v, false, bb.v,
                                                             (short)0, acc[j], false, false);
        }
    }

    // softmax(max - G) == softmax(-G): e_d = exp(Gmin - G_d), normalize per row.
    // Row c = 16*wv + r + 8*(lane>=16) spans acc[0..7][r] across 16 lanes.
    #pragma unroll
    for (int r = 0; r < 8; ++r) {
        float gmin = acc[0][r];
        #pragma unroll
        for (int j = 1; j < 8; ++j) gmin = fminf(gmin, acc[j][r]);
        for (int m = 1; m < 16; m <<= 1) gmin = fminf(gmin, __shfl_xor(gmin, m, 32));
        float e[8], s = 0.f;
        #pragma unroll
        for (int j = 0; j < 8; ++j) { e[j] = __expf(gmin - acc[j][r]); s += e[j]; }
        for (int m = 1; m < 16; m <<= 1) s += __shfl_xor(s, m, 32);
        float rs = 1.f / s;
        int c = (wv << 4) + r + ((lane >> 4) << 3);
        unsigned short* srow = simb + (((size_t)b * Cc + c) << 7);
        #pragma unroll
        for (int j = 0; j < 8; ++j)
            srow[j * 16 + (lane & 15)] = f2bf(e[j] * rs);
    }
}

// ---------------------------------------------------------------------------
// K6: feat = sim @ q (M=128, K=128, N=16384/batch); out = gamma*feat + y.
// Block: 128(c) x 16(n) tile; q tile transposed through LDS for B fragments.
__global__ void k_apply(const unsigned short* __restrict__ simb,
                        const unsigned short* __restrict__ qb,
                        const float* __restrict__ y,
                        const float* __restrict__ gamma,
                        float* __restrict__ out) {
    __shared__ unsigned short Bl[16 * 34];   // [n][d]
    int t = threadIdx.x, lane = t & 31, wv = t >> 5;
    int b   = blockIdx.x >> 10;
    int hwB = (blockIdx.x & 1023) << 4;
    const unsigned short* q = qb + ((size_t)b << 21);

    v8f zero = {};
    v8f acc = zero;
    for (int kk = 0; kk < Cc; kk += 32) {
        __syncthreads();
        {
            int idx = t;
            #pragma unroll
            for (int p = 0; p < 2; ++p) {
                int dl = idx >> 4, nl = idx & 15;
                Bl[nl * 34 + dl] = q[(size_t)(kk + dl) * HWSZ + hwB + nl];
                idx += 256;
            }
        }
        __syncthreads();

        ABfrag a;
        {
            const unsigned short* ar =
                simb + (((size_t)b * Cc + (wv << 4) + (lane & 15)) << 7) + kk + ((lane >> 4) << 4);
            #pragma unroll
            for (int v = 0; v < 8; ++v)
                a.u[v] = *(const unsigned int*)(ar + (v << 1));
        }
        ABfrag bb;
        {
            const unsigned short* br = Bl + (lane & 15) * 34 + ((lane >> 4) << 4);
            #pragma unroll
            for (int v = 0; v < 8; ++v)
                bb.u[v] = *(const unsigned int*)(br + (v << 1));
        }
        acc = __builtin_amdgcn_wmma_f32_16x16x32_bf16(false, a.v, false, bb.v,
                                                      (short)0, acc, false, false);
    }
    float g = gamma[0];
    #pragma unroll
    for (int r = 0; r < 8; ++r) {
        int c = (wv << 4) + r + ((lane >> 4) << 3);
        size_t i = (((size_t)(b * Cc + c)) << 14) + hwB + (lane & 15);
        out[i] = g * acc[r] + y[i];
    }
}

// ---------------------------------------------------------------------------
extern "C" void kernel_launch(void* const* d_in, const int* in_sizes, int n_in,
                              void* d_out, int out_size, void* d_ws, size_t ws_size,
                              hipStream_t stream) {
    (void)in_sizes; (void)n_in; (void)out_size; (void)ws_size;
    const float* x1     = (const float*)d_in[0];
    const float* x2     = (const float*)d_in[1];
    const float* se_w1  = (const float*)d_in[2];
    const float* se_w2  = (const float*)d_in[3];
    const float* conv_w = (const float*)d_in[4];
    const float* bn_g   = (const float*)d_in[5];
    const float* bn_b   = (const float*)d_in[6];
    const float* bn_m   = (const float*)d_in[7];
    const float* bn_v   = (const float*)d_in[8];
    const float* gamma  = (const float*)d_in[9];
    float* out = (float*)d_out;

    char* p = (char*)d_ws;
    float* avg          = (float*)p;          p += (size_t)2048 * 4;
    float* se           = (float*)p;          p += (size_t)2048 * 4;
    float* bnp          = (float*)p;          p += (size_t)256 * 4;
    unsigned short* wbf = (unsigned short*)p; p += (size_t)294912 * 2;
    unsigned short* xh  = (unsigned short*)p; p += (size_t)33554432 * 2;
    float* yb           = (float*)p;          p += (size_t)16777216 * 4;
    unsigned short* qb  = (unsigned short*)p; p += (size_t)16777216 * 2;
    unsigned short* simb= (unsigned short*)p; p += (size_t)131072 * 2;

    k_mean <<<2048,  256, 0, stream>>>(x1, x2, avg);
    k_se   <<<1,     256, 0, stream>>>(avg, se_w1, se_w2, bn_g, bn_b, bn_m, bn_v, se, bnp);
    k_scale<<<32768, 256, 0, stream>>>(x1, x2, se, xh);
    k_wconv<<<288,   256, 0, stream>>>(conv_w, wbf);
    k_conv <<<4096,  256, 0, stream>>>(xh, wbf, bnp, yb, qb);
    k_gram <<<8,     256, 0, stream>>>(qb, simb);
    k_apply<<<8192,  256, 0, stream>>>(simb, qb, yb, gamma, out);
}
